// LSHDecoder_57621281243742
// MI455X (gfx1250) — matrix operand order repaired and checked by reading, not compiled, not run on hardware
//
#include <hip/hip_runtime.h>
#include <hip/hip_bf16.h>
#include <stdint.h>

#define N_PTS 8192
#define DIMS  256
#define BANDS 16
#define ROWS_PER_BAND 8

typedef __attribute__((ext_vector_type(16))) _Float16 v16h;
typedef __attribute__((ext_vector_type(8)))  float    v8f;

union V16H { uint4 u[2]; v16h h; };

// ---------------------------------------------------------------------------
// Kernel 1: per row i of Z (one block per row, 256 threads):
//   - row norm via deterministic LDS tree reduction
//   - write Zn (f16) = Z[i]/||Z[i]||
//   - 128 plane projections (8 waves x 16 planes, coalesced over k)
//   - pack sign bits into 16-byte signature (4 u32 ballots = 16 band-bytes)
// ---------------------------------------------------------------------------
__global__ void lsh_prep_kernel(const float* __restrict__ Z,
                                const float* __restrict__ planes,
                                _Float16* __restrict__ Zn,
                                uint32_t* __restrict__ keys) {
    const int i = blockIdx.x;
    const int t = threadIdx.x;          // 0..255
    const int lane = t & 31;
    const int wave = t >> 5;            // 0..7

    __shared__ float srow[DIMS];        // the row (for projection reuse)
    __shared__ float sred[DIMS];        // reduction / projection results

    float z = Z[(size_t)i * DIMS + t];
    srow[t] = z;
    sred[t] = z * z;
    __syncthreads();

    // deterministic tree reduce for sum of squares
    #pragma unroll
    for (int s = 128; s > 0; s >>= 1) {
        if (t < s) sred[t] += sred[t + s];
        __syncthreads();
    }
    float rn = rsqrtf(sred[0]);
    __syncthreads();                    // everyone read sred[0] before reuse

    Zn[(size_t)i * DIMS + t] = (_Float16)(z * rn);

    // projections: wave w handles planes [16w, 16w+16), coalesced over k
    #pragma unroll 1
    for (int pp = 0; pp < 16; ++pp) {
        int p = wave * 16 + pp;
        const float* pl = planes + (size_t)p * DIMS;
        float acc = 0.f;
        #pragma unroll
        for (int k = lane; k < DIMS; k += 32) acc += pl[k] * srow[k];
        // wave32 butterfly reduce (deterministic)
        #pragma unroll
        for (int off = 16; off; off >>= 1) acc += __shfl_xor(acc, off, 32);
        if (lane == 0) sred[p] = acc;
    }
    __syncthreads();

    // threads 0..127 (waves 0..3, fully converged): ballot sign bits.
    // word w holds planes 32w..32w+31 = bands 4w..4w+3 (byte b, bit r = plane
    // (4w+b)*8+r). Any consistent packing works: keys are only compared.
    if (t < 128) {
        bool bit = sred[t] >= 0.f;
        uint32_t m = (uint32_t)__ballot(bit);
        if (lane == 0) keys[(size_t)i * 4 + (t >> 5)] = m;
    }
}

// exact per-byte nonzero indicator (borrow-free nibble fold)
__device__ __forceinline__ uint32_t nz_bytes(uint32_t t) {
    t = (t | (t >> 4)) & 0x0F0F0F0Fu;
    t = (t | (t >> 2)) & 0x03030303u;
    t = (t | (t >> 1)) & 0x01010101u;
    return t;   // bit0 of each byte == 1 iff that byte was nonzero
}

__device__ __forceinline__ int count_band_matches(uint4 a, uint4 b) {
    int nzc = __popc(nz_bytes(a.x ^ b.x)) + __popc(nz_bytes(a.y ^ b.y)) +
              __popc(nz_bytes(a.z ^ b.z)) + __popc(nz_bytes(a.w ^ b.w));
    return 16 - nzc;    // bands where all 8 sign bits agree
}

// ---------------------------------------------------------------------------
// Kernel 2: sim = Zn @ Zn^T via v_wmma_f32_16x16x32_f16.
// Block = 128 threads (4 waves); wave w -> C tile (blockIdx.y, blockIdx.x*4+w).
// Epilogue: counts * sim on (sim>=0.5 & i!=j), zero elsewhere. Wave-uniform
// skip of the counts path when no element in the tile passes the threshold.
// ---------------------------------------------------------------------------
__global__ void lsh_sim_kernel(const _Float16* __restrict__ Zn,
                               const uint32_t* __restrict__ keys,
                               float* __restrict__ out) {
    const int lane = threadIdx.x & 31;
    const int wave = threadIdx.x >> 5;
    const int rowBase = blockIdx.y * 16;
    const int colBase = (blockIdx.x * 4 + wave) * 16;
    const int lm = lane & 15;
    const int hi = lane >> 4;           // 0 or 1

    // A (16x32 f16, MxK): lanes 0-15 row M=lane, elems 0-7 = K[k0..k0+7],
    //                      elems 8-15 = K[k0+16..k0+23]; lanes 16-31: +8 on K.
    // B (32x16 f16, KxN): lane holds col N=lane%16, same K order as A:
    //                      elems 0-7 = K[k0+8h..], elems 8-15 = K[k0+16+8h..].
    const _Float16* Arow = Zn + (size_t)(rowBase + lm) * DIMS;
    const _Float16* Brow = Zn + (size_t)(colBase + lm) * DIMS;

    v8f c = {};
    #pragma unroll
    for (int k0 = 0; k0 < DIMS; k0 += 32) {
        V16H a, b;
        a.u[0] = *(const uint4*)(Arow + k0 + hi * 8);
        a.u[1] = *(const uint4*)(Arow + k0 + 16 + hi * 8);
        b.u[0] = *(const uint4*)(Brow + k0 + hi * 8);
        b.u[1] = *(const uint4*)(Brow + k0 + 16 + hi * 8);
        c = __builtin_amdgcn_wmma_f32_16x16x32_f16(
                /*neg_a=*/false, a.h, /*neg_b=*/false, b.h,
                /*c_mod=*/(short)0, c, /*reuse_a=*/false, /*reuse_b=*/false);
    }

    // C layout: element v (v8f index) is row M = v + 8*hi, col N = lane&15
    const int j = colBase + lm;
    float* orow = out + (size_t)(rowBase + hi * 8) * N_PTS + j;

    float cv[8];
    bool hit = false;
    #pragma unroll
    for (int v = 0; v < 8; ++v) { cv[v] = c[v]; hit |= (cv[v] >= 0.5f); }

    const uint4* keysv = (const uint4*)keys;
    if (__any(hit)) {
        uint4 kj = keysv[j];
        #pragma unroll
        for (int v = 0; v < 8; ++v) {
            int i = rowBase + hi * 8 + v;
            float s = cv[v];
            float o = 0.f;
            if (s >= 0.5f && i != j) {
                uint4 ki = keysv[i];
                o = (float)count_band_matches(ki, kj) * s;
            }
            orow[(size_t)v * N_PTS] = o;
        }
    } else {
        #pragma unroll
        for (int v = 0; v < 8; ++v) orow[(size_t)v * N_PTS] = 0.f;
    }
}

extern "C" void kernel_launch(void* const* d_in, const int* in_sizes, int n_in,
                              void* d_out, int out_size, void* d_ws, size_t ws_size,
                              hipStream_t stream) {
    const float* Z      = (const float*)d_in[0];   // (8192, 256) f32
    const float* planes = (const float*)d_in[1];   // (128, 256) f32
    float* out = (float*)d_out;                    // (8192, 8192) f32

    _Float16* Zn   = (_Float16*)d_ws;                                   // 4 MB
    uint32_t* keys = (uint32_t*)((char*)d_ws + (size_t)N_PTS * DIMS * 2); // 128 KB

    lsh_prep_kernel<<<N_PTS, 256, 0, stream>>>(Z, planes, Zn, keys);

    dim3 grid(N_PTS / 64, N_PTS / 16);   // 128 x 512 blocks, 4 waves each
    lsh_sim_kernel<<<grid, 128, 0, stream>>>(Zn, keys, out);
}
